// Renderer_45612552684068
// MI455X (gfx1250) — compile-verified, matrix operand order checked by reading, not run on hardware
//
#include <hip/hip_runtime.h>
#include <hip/hip_bf16.h>

typedef _Float16 v16h __attribute__((ext_vector_type(16)));
typedef _Float16 v8h  __attribute__((ext_vector_type(8)));
typedef float    v8f  __attribute__((ext_vector_type(8)));

#define HW    262144   // 512*512
#define WIMG  512
#define NB    4

// A/B fragment load: lane supplies base = row + koff (koff = kc*32 + hi*8).
// Reads halves [0..7] and [16..23] relative to base -> matches ISA 16-bit
// 16x32 A layout (and symmetric B layout) for lanes<16 / lanes>=16.
__device__ __forceinline__ v16h frag_ld(const _Float16* p) {
    v8h lo = *(const v8h*)(p);
    v8h hi = *(const v8h*)(p + 16);
    return __builtin_shufflevector(lo, hi, 0,1,2,3,4,5,6,7,8,9,10,11,12,13,14,15);
}

__device__ __forceinline__ v8f wmma32(v16h a, v16h b, v8f c) {
    return __builtin_amdgcn_wmma_f32_16x16x32_f16(false, a, false, b, (short)0, c, false, false);
}

__device__ __forceinline__ void store8h_relu(_Float16* dst, v8f acc) {
    v8h h;
#pragma unroll
    for (int r = 0; r < 8; ++r) h[r] = (_Float16)fmaxf(acc[r], 0.0f);
    *(v8h*)dst = h;
}

// ---------------------------------------------------------------------------
// Weight prep: f32 -> f16, transposed to [tap][cout][cin] (cin contiguous).
// Segments: w1T 16384 | w2T 4096 | wc1 18432 | wc2 73728 | wc3 110592 | wc4 9216
// ---------------------------------------------------------------------------
__global__ __launch_bounds__(256) void prep_kernel(
    const float* __restrict__ af_w1, const float* __restrict__ af_w2,
    const float* __restrict__ u_k1,  const float* __restrict__ u_k2,
    const float* __restrict__ u_k3,  const float* __restrict__ u_k4,
    _Float16* __restrict__ w1T, _Float16* __restrict__ w2T,
    _Float16* __restrict__ wc1, _Float16* __restrict__ wc2,
    _Float16* __restrict__ wc3, _Float16* __restrict__ wc4)
{
    const int TOT = 16384 + 4096 + 18432 + 73728 + 110592 + 9216;
    for (int i = blockIdx.x * blockDim.x + threadIdx.x; i < TOT; i += gridDim.x * blockDim.x) {
        int j = i;
        if (j < 16384) {                       // w1T[o*128+i] = w1[i*128+o]
            int oo = j >> 7, ii = j & 127;
            w1T[j] = (_Float16)af_w1[ii * 128 + oo];
        } else if ((j -= 16384) < 4096) {      // w2T[o*128+i] = w2[i*32+o]
            int oo = j >> 7, ii = j & 127;
            w2T[j] = (_Float16)af_w2[ii * 32 + oo];
        } else if ((j -= 4096) < 18432) {      // wc1[(tap*64+o)*32+i], cin pad 29->32
            int ii = j & 31, rest = j >> 5, oo = rest & 63, tap = rest >> 6;
            wc1[j] = (ii < 29) ? (_Float16)u_k1[(oo * 29 + ii) * 9 + tap] : (_Float16)0.0f;
        } else if ((j -= 18432) < 73728) {     // wc2[(tap*128+o)*64+i]
            int ii = j & 63, rest = j >> 6, oo = rest & 127, tap = rest >> 7;
            wc2[j] = (_Float16)u_k2[(oo * 64 + ii) * 9 + tap];
        } else if ((j -= 73728) < 110592) {    // wc3[(tap*64+o)*192+i]
            int ii = j % 192, rest = j / 192, oo = rest & 63, tap = rest >> 6;
            wc3[j] = (_Float16)u_k3[(oo * 192 + ii) * 9 + tap];
        } else {                               // wc4[(tap*16+o)*64+i], cout pad 3->16
            j -= 110592;
            int ii = j & 63, rest = j >> 6, oo = rest & 15, tap = rest >> 4;
            wc4[j] = (oo < 3) ? (_Float16)u_k4[(oo * 64 + ii) * 9 + tap] : (_Float16)0.0f;
        }
    }
}

// ---------------------------------------------------------------------------
// MLP: 6 -> 128 -> 128 -> 32, mask, split into x (29ch pad->32, f16 NCHW)
// and rgb_res (3ch, f32 NCHW). 4 waves/block, each wave does 4 tiles of 16 px.
// ---------------------------------------------------------------------------
__global__ __launch_bounds__(128) void mlp_kernel(
    const float* __restrict__ zbuf, const float* __restrict__ ray,
    const float* __restrict__ w0,  const float* __restrict__ b0,
    const float* __restrict__ b1,  const float* __restrict__ b2,
    const _Float16* __restrict__ w1T, const _Float16* __restrict__ w2T,
    _Float16* __restrict__ gx, float* __restrict__ grgb)
{
    __shared__ float s_w0[6 * 128];
    __shared__ float s_b0[128], s_b1[128], s_b2[32];
    __shared__ __attribute__((aligned(16))) _Float16 s_act0[4][16 * 128];
    __shared__ __attribute__((aligned(16))) _Float16 s_act1[4][16 * 128];
    __shared__ float s_mask[4][16];

    const int tid  = threadIdx.x;
    const int w    = tid >> 5;
    const int lane = tid & 31;
    const int hi   = lane >> 4;
    const int ln   = lane & 15;

    for (int i = tid; i < 6 * 128; i += 128) s_w0[i] = w0[i];
    for (int i = tid; i < 128; i += 128) { s_b0[i] = b0[i]; s_b1[i] = b1[i]; }
    if (tid < 32) s_b2[tid] = b2[tid];
    __syncthreads();

    const int P0base = blockIdx.x * 256 + w * 64;

    for (int t = 0; t < 4; ++t) {
        const int P0 = P0base + t * 16;       // 16 contiguous pixels (one W-row segment)
        const int p  = P0 + ln;

        // ---- inputs & layer 0 (K=6, VALU) ----
        float z = zbuf[p];
        float r[7];
#pragma unroll
        for (int j = 0; j < 7; ++j) r[j] = ray[p * 7 + j];
        float f[6];
#pragma unroll
        for (int j = 0; j < 3; ++j) { f[j] = r[j] + r[3 + j] * z / r[6]; f[3 + j] = r[3 + j]; }
        if (hi == 0) s_mask[w][ln] = (z > 0.0f) ? 1.0f : 0.0f;

        const int obase = hi * 64;
        for (int oo = 0; oo < 64; ++oo) {
            float a = s_b0[obase + oo];
#pragma unroll
            for (int j = 0; j < 6; ++j) a += f[j] * s_w0[j * 128 + obase + oo];
            s_act0[w][ln * 128 + obase + oo] = (_Float16)fmaxf(a, 0.0f);
        }
        __syncthreads();

        // ---- layer 1: 128x128 (8 N-tiles x 4 K-chunks of WMMA) ----
        {
            const _Float16* arow = &s_act0[w][ln * 128 + hi * 8];
            for (int nt = 0; nt < 8; ++nt) {
                float bias = s_b1[nt * 16 + ln];
                v8f acc;
#pragma unroll
                for (int rr = 0; rr < 8; ++rr) acc[rr] = bias;
                const _Float16* brow = &w1T[(nt * 16 + ln) * 128 + hi * 8];
#pragma unroll
                for (int kc = 0; kc < 4; ++kc)
                    acc = wmma32(frag_ld(arow + kc * 32), frag_ld(brow + kc * 32), acc);
#pragma unroll
                for (int rr = 0; rr < 8; ++rr)
                    s_act1[w][(rr + hi * 8) * 128 + nt * 16 + ln] = (_Float16)fmaxf(acc[rr], 0.0f);
            }
        }
        __syncthreads();

        // ---- layer 2: 128x32 (2 N-tiles x 4 K-chunks), mask, scatter out ----
        {
            const int b_img = P0 / HW;
            const int pbase = P0 - b_img * HW;
            const _Float16* arow = &s_act1[w][ln * 128 + hi * 8];
            for (int nt = 0; nt < 2; ++nt) {
                float bias = s_b2[nt * 16 + ln];
                v8f acc;
#pragma unroll
                for (int rr = 0; rr < 8; ++rr) acc[rr] = bias;
                const _Float16* brow = &w2T[(nt * 16 + ln) * 128 + hi * 8];
#pragma unroll
                for (int kc = 0; kc < 4; ++kc)
                    acc = wmma32(frag_ld(arow + kc * 32), frag_ld(brow + kc * 32), acc);

                float vals[8];
#pragma unroll
                for (int rr = 0; rr < 8; ++rr) {
                    float mv = s_mask[w][rr + hi * 8];
                    vals[rr] = (mv > 0.0f) ? acc[rr] : 1.0f;
                }
                const int c = nt * 16 + ln;
                if (c < 29) {
                    v8h h;
#pragma unroll
                    for (int rr = 0; rr < 8; ++rr) h[rr] = (_Float16)vals[rr];
                    *(v8h*)&gx[(size_t)(b_img * 32 + c) * HW + pbase + hi * 8] = h;
                } else {
                    v8h zz = {};
                    *(v8h*)&gx[(size_t)(b_img * 32 + c) * HW + pbase + hi * 8] = zz;   // pad ch
#pragma unroll
                    for (int rr = 0; rr < 8; ++rr)
                        grgb[(size_t)(b_img * 3 + (c - 29)) * HW + pbase + rr + hi * 8] = vals[rr];
                }
            }
        }
        __syncthreads();
    }
}

// ---------------------------------------------------------------------------
// conv1: 32(pad) -> 64, 3x3 SAME, relu. Output tile 16x16 per block.
// ---------------------------------------------------------------------------
__global__ __launch_bounds__(256) void conv1_kernel(
    const _Float16* __restrict__ gx, const _Float16* __restrict__ wc1,
    _Float16* __restrict__ ge1)
{
    __shared__ __attribute__((aligned(16))) _Float16 s_in[18 * 18 * 32];
    const int b  = blockIdx.z;
    const int y0 = blockIdx.y * 16, x0 = blockIdx.x * 16;
    const int tid = threadIdx.x;

    for (int i = tid; i < 18 * 18 * 32; i += 256) {
        int c = i / 324, rem = i - c * 324, y = rem / 18, x = rem - y * 18;
        int iy = y0 + y - 1, ix = x0 + x - 1;
        _Float16 v = (_Float16)0.0f;
        if (iy >= 0 && iy < WIMG && ix >= 0 && ix < WIMG)
            v = gx[((size_t)(b * 32 + c) * WIMG + iy) * WIMG + ix];
        s_in[(y * 18 + x) * 32 + c] = v;
    }
    __syncthreads();

    const int lane = tid & 31, wid = tid >> 5, hi = lane >> 4, ln = lane & 15;
    for (int u = wid; u < 64; u += 8) {
        const int yr = u & 15, nt = u >> 4;
        v8f acc = {};
        const _Float16* brow = wc1 + (nt * 16 + ln) * 32 + hi * 8;
#pragma unroll
        for (int tap = 0; tap < 9; ++tap) {
            const int dy = tap / 3, dx = tap - dy * 3;
            const _Float16* ap = &s_in[((yr + dy) * 18 + ln + dx) * 32 + hi * 8];
            acc = wmma32(frag_ld(ap), frag_ld(brow + tap * 64 * 32), acc);
        }
        store8h_relu(&ge1[((size_t)(b * 64 + nt * 16 + ln) * WIMG + (y0 + yr)) * WIMG + x0 + hi * 8], acc);
    }
}

// ---------------------------------------------------------------------------
// conv2: 64 -> 128, 3x3 stride-2 SAME, relu. Output tile 4x16 (on 256x256).
// ---------------------------------------------------------------------------
__global__ __launch_bounds__(256) void conv2_kernel(
    const _Float16* __restrict__ ge1, const _Float16* __restrict__ wc2,
    _Float16* __restrict__ gd)
{
    __shared__ __attribute__((aligned(16))) _Float16 s_in[9 * 34 * 64];
    const int b   = blockIdx.z;
    const int oy0 = blockIdx.y * 4, ox0 = blockIdx.x * 16;
    const int iny0 = oy0 * 2 - 1, inx0 = ox0 * 2 - 1;
    const int tid = threadIdx.x;

    for (int i = tid; i < 9 * 34 * 64; i += 256) {
        int c = i / 306, rem = i - c * 306, y = rem / 34, x = rem - y * 34;
        int iy = iny0 + y, ix = inx0 + x;
        _Float16 v = (_Float16)0.0f;
        if (iy >= 0 && iy < WIMG && ix >= 0 && ix < WIMG)
            v = ge1[((size_t)(b * 64 + c) * WIMG + iy) * WIMG + ix];
        s_in[(y * 34 + x) * 64 + c] = v;
    }
    __syncthreads();

    const int lane = tid & 31, wid = tid >> 5, hi = lane >> 4, ln = lane & 15;
    for (int u = wid; u < 32; u += 8) {
        const int yr = u & 3, nt = u >> 2;
        v8f acc = {};
        const _Float16* brow = wc2 + (nt * 16 + ln) * 64 + hi * 8;
#pragma unroll
        for (int tap = 0; tap < 9; ++tap) {
            const int dy = tap / 3, dx = tap - dy * 3;
            const _Float16* ap = &s_in[((2 * yr + dy) * 34 + 2 * ln + dx) * 64 + hi * 8];
            const _Float16* bp = brow + tap * 128 * 64;
#pragma unroll
            for (int kc = 0; kc < 2; ++kc)
                acc = wmma32(frag_ld(ap + kc * 32), frag_ld(bp + kc * 32), acc);
        }
        store8h_relu(&gd[((size_t)(b * 128 + nt * 16 + ln) * 256 + (oy0 + yr)) * 256 + ox0 + hi * 8], acc);
    }
}

// ---------------------------------------------------------------------------
// conv3: concat(upsample(d), e1) = 192 -> 64, 3x3 SAME, relu.
// Upsample + concat folded into LDS gather. Output tile 4x16.
// ---------------------------------------------------------------------------
__global__ __launch_bounds__(256) void conv3_kernel(
    const _Float16* __restrict__ gd, const _Float16* __restrict__ ge1,
    const _Float16* __restrict__ wc3, _Float16* __restrict__ gm)
{
    __shared__ __attribute__((aligned(16))) _Float16 s_in[6 * 18 * 192];
    const int b  = blockIdx.z;
    const int y0 = blockIdx.y * 4, x0 = blockIdx.x * 16;
    const int tid = threadIdx.x;

    for (int i = tid; i < 6 * 18 * 192; i += 256) {
        int c = i / 108, rem = i - c * 108, y = rem / 18, x = rem - y * 18;
        int iy = y0 + y - 1, ix = x0 + x - 1;
        _Float16 v = (_Float16)0.0f;
        if (iy >= 0 && iy < WIMG && ix >= 0 && ix < WIMG) {
            if (c < 128) v = gd[((size_t)(b * 128 + c) * 256 + (iy >> 1)) * 256 + (ix >> 1)];
            else         v = ge1[((size_t)(b * 64 + (c - 128)) * WIMG + iy) * WIMG + ix];
        }
        s_in[(y * 18 + x) * 192 + c] = v;
    }
    __syncthreads();

    const int lane = tid & 31, wid = tid >> 5, hi = lane >> 4, ln = lane & 15;
    for (int u = wid; u < 16; u += 8) {
        const int yr = u & 3, nt = u >> 2;
        v8f acc = {};
        const _Float16* brow = wc3 + (nt * 16 + ln) * 192 + hi * 8;
#pragma unroll
        for (int tap = 0; tap < 9; ++tap) {
            const int dy = tap / 3, dx = tap - dy * 3;
            const _Float16* ap = &s_in[((yr + dy) * 18 + ln + dx) * 192 + hi * 8];
            const _Float16* bp = brow + tap * 64 * 192;
#pragma unroll
            for (int kc = 0; kc < 6; ++kc)
                acc = wmma32(frag_ld(ap + kc * 32), frag_ld(bp + kc * 32), acc);
        }
        store8h_relu(&gm[((size_t)(b * 64 + nt * 16 + ln) * WIMG + (y0 + yr)) * WIMG + x0 + hi * 8], acc);
    }
}

// ---------------------------------------------------------------------------
// conv4: 64 -> 3 (padded to 16 couts), 3x3 SAME, + rgb_res. f32 output.
// Output tile 16x16.
// ---------------------------------------------------------------------------
__global__ __launch_bounds__(256) void conv4_kernel(
    const _Float16* __restrict__ gm, const _Float16* __restrict__ wc4,
    const float* __restrict__ grgb, float* __restrict__ out)
{
    __shared__ __attribute__((aligned(16))) _Float16 s_in[18 * 18 * 64];
    const int b  = blockIdx.z;
    const int y0 = blockIdx.y * 16, x0 = blockIdx.x * 16;
    const int tid = threadIdx.x;

    for (int i = tid; i < 18 * 18 * 64; i += 256) {
        int c = i / 324, rem = i - c * 324, y = rem / 18, x = rem - y * 18;
        int iy = y0 + y - 1, ix = x0 + x - 1;
        _Float16 v = (_Float16)0.0f;
        if (iy >= 0 && iy < WIMG && ix >= 0 && ix < WIMG)
            v = gm[((size_t)(b * 64 + c) * WIMG + iy) * WIMG + ix];
        s_in[(y * 18 + x) * 64 + c] = v;
    }
    __syncthreads();

    const int lane = tid & 31, wid = tid >> 5, hi = lane >> 4, ln = lane & 15;
    for (int u = wid; u < 16; u += 8) {
        const int yr = u;
        v8f acc = {};
        const _Float16* brow = wc4 + ln * 64 + hi * 8;
#pragma unroll
        for (int tap = 0; tap < 9; ++tap) {
            const int dy = tap / 3, dx = tap - dy * 3;
            const _Float16* ap = &s_in[((yr + dy) * 18 + ln + dx) * 64 + hi * 8];
            const _Float16* bp = brow + tap * 16 * 64;
#pragma unroll
            for (int kc = 0; kc < 2; ++kc)
                acc = wmma32(frag_ld(ap + kc * 32), frag_ld(bp + kc * 32), acc);
        }
        if (ln < 3) {   // only real output channels; after all WMMAs (EXEC safe)
            size_t idx = ((size_t)(b * 3 + ln) * WIMG + (y0 + yr)) * WIMG + x0 + hi * 8;
#pragma unroll
            for (int rr = 0; rr < 8; ++rr) out[idx + rr] = acc[rr] + grgb[idx + rr];
        }
    }
}

// ---------------------------------------------------------------------------
extern "C" void kernel_launch(void* const* d_in, const int* in_sizes, int n_in,
                              void* d_out, int out_size, void* d_ws, size_t ws_size,
                              hipStream_t stream) {
    const float* zbuf  = (const float*)d_in[0];
    const float* ray   = (const float*)d_in[1];
    const float* af_w0 = (const float*)d_in[4];
    const float* af_b0 = (const float*)d_in[5];
    const float* af_w1 = (const float*)d_in[6];
    const float* af_b1 = (const float*)d_in[7];
    const float* af_w2 = (const float*)d_in[8];
    const float* af_b2 = (const float*)d_in[9];
    const float* u_k1  = (const float*)d_in[10];
    const float* u_k2  = (const float*)d_in[11];
    const float* u_k3  = (const float*)d_in[12];
    const float* u_k4  = (const float*)d_in[13];
    float* out = (float*)d_out;

    char* ws = (char*)d_ws;
    size_t off = 0;
    auto take = [&](size_t bytes) -> char* {
        char* p = ws + off;
        off = (off + bytes + 255) & ~(size_t)255;
        return p;
    };
    _Float16* w1T = (_Float16*)take((size_t)16384 * 2);
    _Float16* w2T = (_Float16*)take((size_t)4096 * 2);
    _Float16* wc1 = (_Float16*)take((size_t)18432 * 2);
    _Float16* wc2 = (_Float16*)take((size_t)73728 * 2);
    _Float16* wc3 = (_Float16*)take((size_t)110592 * 2);
    _Float16* wc4 = (_Float16*)take((size_t)9216 * 2);
    _Float16* gx  = (_Float16*)take((size_t)NB * 32 * HW * 2);   // padded fmap[:, :29]
    float*    grgb = (float*)   take((size_t)NB * 3 * HW * 4);   // fmap[:, 29:32]
    _Float16* ge1 = (_Float16*)take((size_t)NB * 64 * HW * 2);
    _Float16* gd  = (_Float16*)take((size_t)NB * 128 * 65536 * 2);
    _Float16* gm  = (_Float16*)take((size_t)NB * 64 * HW * 2);

    prep_kernel<<<256, 256, 0, stream>>>(af_w1, af_w2, u_k1, u_k2, u_k3, u_k4,
                                         w1T, w2T, wc1, wc2, wc3, wc4);
    mlp_kernel<<<4096, 128, 0, stream>>>(zbuf, ray, af_w0, af_b0, af_b1, af_b2,
                                         w1T, w2T, gx, grgb);
    conv1_kernel<<<dim3(32, 32, NB), 256, 0, stream>>>(gx, wc1, ge1);
    conv2_kernel<<<dim3(16, 64, NB), 256, 0, stream>>>(ge1, wc2, gd);
    conv3_kernel<<<dim3(32, 128, NB), 256, 0, stream>>>(gd, ge1, wc3, gm);
    conv4_kernel<<<dim3(32, 32, NB), 256, 0, stream>>>(gm, wc4, grgb, out);
}
